// HiModel_54726473285909
// MI455X (gfx1250) — matrix operand order repaired and checked by reading, not compiled, not run on hardware
//
#include <hip/hip_runtime.h>
#include <hip/hip_bf16.h>

typedef __bf16 bf16_t;
typedef __attribute__((ext_vector_type(16))) __bf16 v16bf;
typedef __attribute__((ext_vector_type(8)))  float  v8f;
typedef __attribute__((ext_vector_type(4)))  float  f32x4;

#define EPSBN 1e-5f

// ---------------------------------------------------------------------------
// selu(z) = scale * (z > 0 ? z : alpha*(exp(z)-1))
__device__ __forceinline__ float selu_dev(float z) {
    const float sc = 1.0507009873554805f;
    const float al = 1.6732632423543772f;
    return z > 0.f ? sc * z : sc * al * (__expf(z) - 1.f);
}

// ---------------------------------------------------------------------------
// Transpose fp32 [K,Nn] weight into bf16 [Nn,K] (K contiguous per row) so B
// WMMA fragments are contiguous 16xbf16 loads per lane.
__global__ void convert_wt_kernel(const float* __restrict__ W, bf16_t* __restrict__ Wt,
                                  int K, int Nn) {
    size_t idx = (size_t)blockIdx.x * blockDim.x + threadIdx.x;
    size_t total = (size_t)K * Nn;
    if (idx >= total) return;
    size_t n  = idx / (size_t)K;
    size_t kk = idx % (size_t)K;
    Wt[idx] = (bf16_t)W[kk * (size_t)Nn + n];
}

__global__ void zero_kernel(float* __restrict__ p, size_t n) {
    size_t i = (size_t)blockIdx.x * blockDim.x + threadIdx.x;
    if (i < n) p[i] = 0.f;
}

// ---------------------------------------------------------------------------
// deg[dst] += 1 per edge (hardware fp32 atomic, L2-resident since N*4=200KB)
__global__ void deg_kernel(const long long* __restrict__ dst, float* __restrict__ deg,
                           long long E) {
    long long i = (long long)blockIdx.x * blockDim.x + threadIdx.x;
    if (i < E)
        __hip_atomic_fetch_add(&deg[dst[i]], 1.0f, __ATOMIC_RELAXED, __HIP_MEMORY_SCOPE_AGENT);
}

__global__ void finalize_deg_kernel(const float* __restrict__ deg, float* __restrict__ dinv,
                                    float* __restrict__ invdeg, int N) {
    int i = blockIdx.x * blockDim.x + threadIdx.x;
    if (i >= N) return;
    float d = deg[i] + 1.f;     // +1 self loop
    dinv[i]   = rsqrtf(d);
    invdeg[i] = 1.f / d;
}

// ---------------------------------------------------------------------------
// WMMA GEMM: C[M,Nn] = epilogue( A[M,K] (opt. BN per-column) x Bt[Nn,K]^T )
// Block = 256 threads = 8 waves; each wave computes a 16(M) x (16*NT)(N) strip
// -> block tile is 128 x (16*NT). All NT B fragments are preloaded before the
// WMMA chain so s_wait_loadcnt staggers instead of fully serializing.
// EPI 0: plain store (h = xW, bias added later with self-loop term)
// EPI 1: bias + soft-threshold(SELU) + 0.5*lm + 0.5*st   (GEMM1 fusion)
template <int EPI, bool HAS_BN, int NT>
__global__ __launch_bounds__(256)
void gemm_wmma_kernel(const float* __restrict__ A,
                      const float* __restrict__ gamma, const float* __restrict__ beta,
                      const bf16_t* __restrict__ Bt, const float* __restrict__ bias,
                      const float* __restrict__ lm, const float* __restrict__ theta,
                      float* __restrict__ out, int M, int K, int Nn) {
    const int lane = threadIdx.x & 31;
    const int wave = threadIdx.x >> 5;
    const int half = lane >> 4;       // 0 or 1
    const int l16  = lane & 15;

    // A-fragment row for this lane (16-bit A 16x32 layout: lane holds row M=l16)
    int rowA = blockIdx.x * 128 + wave * 16 + l16;
    if (rowA >= M) rowA = M - 1;      // clamp loads, predicate stores (EXEC stays all-1s)
    const float* Arow = A + (size_t)rowA * K;

    const int tileN = blockIdx.y * (16 * NT);
    const float bnrs = rsqrtf(1.f + EPSBN);

    v8f acc[NT];
#pragma unroll
    for (int i = 0; i < NT; ++i)
#pragma unroll
        for (int j = 0; j < 8; ++j) acc[i][j] = 0.f;

    for (int k0 = 0; k0 < K; k0 += 32) {
        // A 16x32 bf16 fragment: lane holds K=[k0+half*16, +16) contiguous
        const int c0 = k0 + half * 16;
        const int c1 = k0 + half * 16 + 8;
        f32x4 a0 = *(const f32x4*)(Arow + c0);
        f32x4 a1 = *(const f32x4*)(Arow + c0 + 4);
        f32x4 a2 = *(const f32x4*)(Arow + c1);
        f32x4 a3 = *(const f32x4*)(Arow + c1 + 4);

        // B 32x16 bf16 fragments: lane holds column n=l16, K=[k0+half*16, +16).
        // Preload ALL NT fragments so the WMMA chain overlaps the loads.
        const bf16_t* Bbase = Bt + (size_t)(tileN + l16) * K + k0 + half * 16;
        v16bf bfrag[NT];
#pragma unroll
        for (int nt = 0; nt < NT; ++nt)
            bfrag[nt] = *(const v16bf*)(Bbase + (size_t)nt * 16 * K);

        if (HAS_BN) {
            f32x4 g0 = *(const f32x4*)(gamma + c0);
            f32x4 g1 = *(const f32x4*)(gamma + c0 + 4);
            f32x4 g2 = *(const f32x4*)(gamma + c1);
            f32x4 g3 = *(const f32x4*)(gamma + c1 + 4);
            f32x4 b0v = *(const f32x4*)(beta + c0);
            f32x4 b1v = *(const f32x4*)(beta + c0 + 4);
            f32x4 b2v = *(const f32x4*)(beta + c1);
            f32x4 b3v = *(const f32x4*)(beta + c1 + 4);
            a0 = a0 * (g0 * bnrs) + b0v;
            a1 = a1 * (g1 * bnrs) + b1v;
            a2 = a2 * (g2 * bnrs) + b2v;
            a3 = a3 * (g3 * bnrs) + b3v;
        }
        v16bf af;
#pragma unroll
        for (int j = 0; j < 4; ++j) {
            af[j]      = (bf16_t)a0[j];
            af[4 + j]  = (bf16_t)a1[j];
            af[8 + j]  = (bf16_t)a2[j];
            af[12 + j] = (bf16_t)a3[j];
        }

        if (k0 + 32 < K) __builtin_prefetch(Arow + k0 + 32, 0, 3);  // global_prefetch_b8

#pragma unroll
        for (int nt = 0; nt < NT; ++nt)
            acc[nt] = __builtin_amdgcn_wmma_f32_16x16x32_bf16(
                false, af, false, bfrag[nt], (short)0, acc[nt], false, false);
    }

    // C/D layout: lane holds column n=l16, VGPR r holds row r + 8*half
    const float th = (EPI == 1) ? theta[0] : 0.f;
#pragma unroll
    for (int nt = 0; nt < NT; ++nt) {
        const int col = tileN + nt * 16 + l16;
#pragma unroll
        for (int r = 0; r < 8; ++r) {
            const int grow = blockIdx.x * 128 + wave * 16 + r + 8 * half;
            if (grow < M) {
                float v = acc[nt][r];
                if (EPI == 1) {
                    v += bias[col];
                    float st = selu_dev(v - th) - selu_dev(-v - th);
                    float lmv = lm[(size_t)grow * Nn + col];
                    v = 0.5f * lmv + 0.5f * st;   // LAMB = 0.5
                }
                out[(size_t)grow * Nn + col] = v;
            }
        }
    }
}

// ---------------------------------------------------------------------------
// One block per edge: out[dst,:] += (dinv[src]*dinv[dst]) * h[src,:]
// Dest buffer (<=102 MB) is L2-resident -> fp32 atomics resolve in L2.
__global__ void aggregate_kernel(const float* __restrict__ h,
                                 const long long* __restrict__ src,
                                 const long long* __restrict__ dst,
                                 const float* __restrict__ dinv,
                                 float* __restrict__ out, int F) {
    __shared__ float scoef;
    __shared__ long long ssrc, sdst;
    if (threadIdx.x == 0) {
        long long s = src[blockIdx.x];
        long long d = dst[blockIdx.x];
        ssrc = s; sdst = d;
        scoef = dinv[s] * dinv[d];
    }
    __syncthreads();
    const float coef = scoef;
    const float* hs = h + (size_t)ssrc * F;
    float* od = out + (size_t)sdst * F;
    for (int f = threadIdx.x; f < F; f += blockDim.x)
        __hip_atomic_fetch_add(&od[f], hs[f] * coef, __ATOMIC_RELAXED, __HIP_MEMORY_SCOPE_AGENT);
}

// ---------------------------------------------------------------------------
// out = agg + h * invdeg[row] + bias, optionally BN(eval)+ReLU fused
template <bool BNRELU>
__global__ void selfloop_kernel(const float* __restrict__ agg, const float* __restrict__ h,
                                const float* __restrict__ invdeg, const float* __restrict__ bias,
                                const float* __restrict__ bng, const float* __restrict__ bnb,
                                float* __restrict__ out, size_t total, int fshift) {
    size_t idx = (size_t)blockIdx.x * blockDim.x + threadIdx.x;
    if (idx >= total) return;
    const int F = 1 << fshift;
    size_t i = idx >> fshift;
    int    f = (int)(idx & (size_t)(F - 1));
    float v = agg[idx] + h[idx] * invdeg[i] + bias[f];
    if (BNRELU) {
        v = v * (bng[f] * rsqrtf(1.f + EPSBN)) + bnb[f];
        v = fmaxf(v, 0.f);
    }
    out[idx] = v;
}

// ---------------------------------------------------------------------------
extern "C" void kernel_launch(void* const* d_in, const int* in_sizes, int n_in,
                              void* d_out, int out_size, void* d_ws, size_t ws_size,
                              hipStream_t stream) {
    (void)n_in; (void)out_size; (void)ws_size;
    const float*     lm_emb = (const float*)d_in[0];
    const float*     fea    = (const float*)d_in[1];
    const long long* edge   = (const long long*)d_in[2];
    const float*     s_gamma= (const float*)d_in[3];
    const float*     s_beta = (const float*)d_in[4];
    const float*     s_W    = (const float*)d_in[5];
    const float*     s_b    = (const float*)d_in[6];
    const float*     theta  = (const float*)d_in[7];
    const float*     W0     = (const float*)d_in[8];
    const float*     b0     = (const float*)d_in[9];
    const float*     W1     = (const float*)d_in[10];
    const float*     b1     = (const float*)d_in[11];
    const float*     W2     = (const float*)d_in[12];
    const float*     b2     = (const float*)d_in[13];
    const float*     bn0g   = (const float*)d_in[14];
    const float*     bn0b   = (const float*)d_in[15];
    const float*     bn1g   = (const float*)d_in[16];
    const float*     bn1b   = (const float*)d_in[17];

    const int  N = in_sizes[0] / 768;
    const long long E = (long long)in_sizes[2] / 2;
    const long long* esrc = edge;
    const long long* edst = edge + E;

    // workspace arena
    size_t off = 0;
    char* base = (char*)d_ws;
    auto arena = [&](size_t bytes) -> void* {
        void* p = base + off;
        off += (bytes + 255) & ~(size_t)255;
        return p;
    };
    float*  deg    = (float*) arena((size_t)N * 4);
    float*  dinv   = (float*) arena((size_t)N * 4);
    float*  invdeg = (float*) arena((size_t)N * 4);
    bf16_t* sWt    = (bf16_t*)arena((size_t)2 * 2048 * 768 * 2);
    bf16_t* W0t    = (bf16_t*)arena((size_t)768 * 512 * 2);
    bf16_t* W1t    = (bf16_t*)arena((size_t)512 * 512 * 2);
    bf16_t* W2t    = (bf16_t*)arena((size_t)512 * 64 * 2);
    float*  xbuf   = (float*) arena((size_t)N * 768 * 4);
    float*  hbuf   = (float*) arena((size_t)N * 512 * 4);
    float*  aggbuf = (float*) arena((size_t)N * 512 * 4);
    float*  out    = (float*)d_out;

    auto cdiv = [](size_t a, size_t b) { return (unsigned)((a + b - 1) / b); };

    // -------- weights -> bf16, transposed (once) --------
    convert_wt_kernel<<<cdiv((size_t)2048 * 768, 256), 256, 0, stream>>>(s_W, sWt, 2048, 768);
    convert_wt_kernel<<<cdiv((size_t)2048 * 768, 256), 256, 0, stream>>>(
        s_W + (size_t)2048 * 768, sWt + (size_t)2048 * 768, 2048, 768);
    convert_wt_kernel<<<cdiv((size_t)768 * 512, 256), 256, 0, stream>>>(W0, W0t, 768, 512);
    convert_wt_kernel<<<cdiv((size_t)512 * 512, 256), 256, 0, stream>>>(W1, W1t, 512, 512);
    convert_wt_kernel<<<cdiv((size_t)512 * 64, 256), 256, 0, stream>>>(W2, W2t, 512, 64);

    // -------- degrees --------
    zero_kernel<<<cdiv((size_t)N, 256), 256, 0, stream>>>(deg, (size_t)N);
    deg_kernel<<<cdiv((size_t)E, 256), 256, 0, stream>>>(edst, deg, E);
    finalize_deg_kernel<<<cdiv((size_t)N, 256), 256, 0, stream>>>(deg, dinv, invdeg, N);

    const unsigned nb128 = cdiv((size_t)N, 128);

    for (int k = 0; k < 2; ++k) {
        // GEMM1: x = 0.5*lm + 0.5*softthr(BN(fea[k]) @ s_W[k] + s_b[k])  [N,768]
        gemm_wmma_kernel<1, true, 8><<<dim3(nb128, 6), 256, 0, stream>>>(
            fea + (size_t)k * N * 2048, s_gamma, s_beta,
            sWt + (size_t)k * 2048 * 768, s_b + (size_t)k * 768,
            lm_emb, theta, xbuf, N, 2048, 768);

        // ---- GCN layer 0: h = x @ W0 ; agg ; x1 = relu(bn0(agg + h/deg + b0)) ----
        gemm_wmma_kernel<0, false, 8><<<dim3(nb128, 4), 256, 0, stream>>>(
            xbuf, nullptr, nullptr, W0t, nullptr, nullptr, nullptr, hbuf, N, 768, 512);
        zero_kernel<<<cdiv((size_t)N * 512, 256), 256, 0, stream>>>(aggbuf, (size_t)N * 512);
        aggregate_kernel<<<(unsigned)E, 256, 0, stream>>>(hbuf, esrc, edst, dinv, aggbuf, 512);
        selfloop_kernel<true><<<cdiv((size_t)N * 512, 256), 256, 0, stream>>>(
            aggbuf, hbuf, invdeg, b0, bn0g, bn0b, aggbuf, (size_t)N * 512, 9);  // in-place -> x1

        // ---- GCN layer 1: h = x1 @ W1 ; agg ; final[k] = relu(bn1(...)) ----
        gemm_wmma_kernel<0, false, 8><<<dim3(nb128, 4), 256, 0, stream>>>(
            aggbuf, nullptr, nullptr, W1t, nullptr, nullptr, nullptr, hbuf, N, 512, 512);
        zero_kernel<<<cdiv((size_t)N * 512, 256), 256, 0, stream>>>(aggbuf, (size_t)N * 512);
        aggregate_kernel<<<(unsigned)E, 256, 0, stream>>>(hbuf, esrc, edst, dinv, aggbuf, 512);
        float* xfinal = out + (size_t)k * N * 512;
        selfloop_kernel<true><<<cdiv((size_t)N * 512, 256), 256, 0, stream>>>(
            aggbuf, hbuf, invdeg, b1, bn1g, bn1b, xfinal, (size_t)N * 512, 9);

        // ---- prediction head: h = final[k] @ W2 ; agg ; final_pred[k] ----
        gemm_wmma_kernel<0, false, 4><<<dim3(nb128, 1), 256, 0, stream>>>(
            xfinal, nullptr, nullptr, W2t, nullptr, nullptr, nullptr, hbuf, N, 512, 64);
        zero_kernel<<<cdiv((size_t)N * 64, 256), 256, 0, stream>>>(aggbuf, (size_t)N * 64);
        aggregate_kernel<<<(unsigned)E, 64, 0, stream>>>(hbuf, esrc, edst, dinv, aggbuf, 64);
        selfloop_kernel<false><<<cdiv((size_t)N * 64, 256), 256, 0, stream>>>(
            aggbuf, hbuf, invdeg, b2, nullptr, nullptr,
            out + (size_t)2 * N * 512 + (size_t)k * N * 64, (size_t)N * 64, 6);
    }
}
